// WaveMambaBlock_6511170421682
// MI455X (gfx1250) — matrix operand order repaired and checked by reading, not compiled, not run on hardware
//
#include <hip/hip_runtime.h>

typedef _Float16 f16;
typedef __attribute__((ext_vector_type(16))) _Float16 v16h;
typedef __attribute__((ext_vector_type(8)))  float    v8f;

#define B_    8
#define CIN   64
#define T_    128
#define F_    32
#define T2    64
#define F2    16
#define L_    1024
#define MROWS 8192      // B_ * L_
#define MAIN_ 128
#define AUX_  64
#define DIN   256
#define NST   16
#define DTR   8
#define FUSED 192
#define SQ2f  0.70710678f
#define BNS   0.99999500003749971f   // 1/sqrt(1+1e-5)

__device__ __forceinline__ float sigm_(float x)    { return 1.f / (1.f + __expf(-x)); }
__device__ __forceinline__ float silu_(float x)    { return x * sigm_(x); }
__device__ __forceinline__ float softplus_(float x){ return (x > 20.f) ? x : log1pf(__expf(x)); }

// ---------------------------------------------------------------------------
// WMMA fragment helpers (layouts per CDNA5 ISA 7.12.2, wave32)
// ---------------------------------------------------------------------------
// A row-major [M x K]; lane's row = m0+(lane&15), K base = k0 + 8*(lane>=16).
// Elements 0..7 -> K=base+0..7, elements 8..15 -> K=base+16..23 (two b128s).
__device__ __forceinline__ v16h load_a16(const f16* __restrict__ p) {
  v16h a;
#pragma unroll
  for (int e = 0; e < 8; ++e)  a[e] = p[e];
#pragma unroll
  for (int e = 8; e < 16; ++e) a[e] = p[e + 8];
  return a;
}

// GEMM: C[M,N](f32) = A[M,K](f16 row-major) * Bsw (f16, fragment-swizzled).
// Bsw layout: tile (nt,kt) at (nt*(K/32)+kt)*512, within tile lane*16+e
// holding B[kt*32 + 8*(lane>=16) + (e&7) + 16*(e>>3)][nt*16 + (lane&15)].
// Each wave computes 4 consecutive 16x16 M-tiles sharing one B fragment.
// K, N compile-time -> all strides fold into immediate offsets.
template<int K, int N>
__global__ void gemm_sw(const f16* __restrict__ A, const f16* __restrict__ Bsw,
                        float* __restrict__ C) {
  constexpr int KT = K >> 5;
  constexpr int TN = N >> 4;
  int lane = threadIdx.x & 31;
  int wave = blockIdx.x * (blockDim.x >> 5) + (threadIdx.x >> 5);
  int mg = wave / TN, nt = wave % TN;
  int m0 = mg << 6;
  int n0 = nt << 4;

  const f16* ap = A + (m0 + (lane & 15)) * K + ((lane >> 4) << 3);
  const f16* bp = Bsw + nt * (KT * 512) + lane * 16;
  float*     cp = C + (m0 + ((lane >> 4) << 3)) * N + n0 + (lane & 15);

  v8f acc0 = {}, acc1 = {}, acc2 = {}, acc3 = {};
#pragma unroll
  for (int kt = 0; kt < KT; ++kt) {
    if (kt + 1 < KT) __builtin_prefetch(bp + (kt + 1) * 512, 0, 1);
    v16h b  = *(const v16h*)(bp + kt * 512);
    v16h a0 = load_a16(ap + kt * 32);
    v16h a1 = load_a16(ap + kt * 32 + 16 * K);
    v16h a2 = load_a16(ap + kt * 32 + 32 * K);
    v16h a3 = load_a16(ap + kt * 32 + 48 * K);
    acc0 = __builtin_amdgcn_wmma_f32_16x16x32_f16(false, a0, false, b, (short)0, acc0, false, false);
    acc1 = __builtin_amdgcn_wmma_f32_16x16x32_f16(false, a1, false, b, (short)0, acc1, false, false);
    acc2 = __builtin_amdgcn_wmma_f32_16x16x32_f16(false, a2, false, b, (short)0, acc2, false, false);
    acc3 = __builtin_amdgcn_wmma_f32_16x16x32_f16(false, a3, false, b, (short)0, acc3, false, false);
  }
#pragma unroll
  for (int r = 0; r < 8; ++r) {
    cp[r * N]          = acc0[r];
    cp[r * N + 16 * N] = acc1[r];
    cp[r * N + 32 * N] = acc2[r];
    cp[r * N + 48 * N] = acc3[r];
  }
}

// ---------------------------------------------------------------------------
// Weight prep: swizzle (optionally transpose) + zero-pad into fragment order.
// transB=1: W is [N x K] row-major (take W[n*K+k]); transB=0: W is [K x N].
// ---------------------------------------------------------------------------
__global__ void k_swizzle_w(const float* __restrict__ W, f16* __restrict__ Bsw,
                            int N, int K, int Np, int Kp, int transB) {
  int i = blockIdx.x * blockDim.x + threadIdx.x;
  int KT = Kp >> 5;
  int total = (Np >> 4) * KT * 512;
  if (i >= total) return;
  int e    = i & 15;
  int lane = (i >> 4) & 31;
  int tile = i >> 9;
  int nt = tile / KT, kt = tile % KT;
  int k = kt * 32 + ((lane >> 4) << 3) + (e & 7) + ((e >> 3) << 4);
  int n = (nt << 4) + (lane & 15);
  float v = 0.f;
  if (k < K && n < N) v = transB ? W[(size_t)n * K + k] : W[(size_t)k * N + n];
  Bsw[i] = (f16)v;
}

// ---------------------------------------------------------------------------
// Stage kernels
// ---------------------------------------------------------------------------
// Haar wavelet + im2col for patch projections: rows r=(b,t,f), cols c*2+p
__global__ void k_wavelet(const float* __restrict__ x,
                          f16* __restrict__ Alow, f16* __restrict__ Ahigh) {
  int i = blockIdx.x * blockDim.x + threadIdx.x;
  if (i >= MROWS * 128) return;
  int col = i & 127, r = i >> 7;
  int p = col & 1, c = col >> 1;
  int f = r & 15, t = (r >> 4) & 63, b = r >> 10;
  int ff = 2 * f + p;
  const float* xb = x + (((size_t)(b * CIN + c) * T_ + 2 * t) * F_ + ff);
  float xe = xb[0], xo = xb[F_];
  Alow[i]  = (f16)((xe + xo) * SQ2f);
  Ahigh[i] = (f16)((xo - xe) * SQ2f);
}

// relu((Y+bias)*g*BNS + beta) -> mainseq [row,(c=128)] == x_seq layout
__global__ void k_patch_post_main(const float* __restrict__ Y, const float* __restrict__ bias,
                                  const float* __restrict__ g, const float* __restrict__ beta,
                                  float* __restrict__ mainseq) {
  int i = blockIdx.x * blockDim.x + threadIdx.x;
  if (i >= MROWS * MAIN_) return;
  int o = i & 127;
  float v = (Y[i] + bias[o]) * (g[o] * BNS) + beta[o];
  mainseq[i] = fmaxf(v, 0.f);
}

// relu((Y+bias)*g*BNS + beta) -> auxin (b,64,64,16)
__global__ void k_patch_post_aux(const float* __restrict__ Y, const float* __restrict__ bias,
                                 const float* __restrict__ g, const float* __restrict__ beta,
                                 float* __restrict__ auxin) {
  int i = blockIdx.x * blockDim.x + threadIdx.x;
  if (i >= MROWS * AUX_) return;
  int o = i & 63, r = i >> 6;
  int f = r & 15, t = (r >> 4) & 63, b = r >> 10;
  float v = fmaxf((Y[i] + bias[o]) * (g[o] * BNS) + beta[o], 0.f);
  auxin[((size_t)(b * AUX_ + o) * T2 + t) * F2 + f] = v;
}

// LayerNorm over 128 channels; one wave per row. Writes forward + L-reversed f16.
__global__ void k_layernorm(const float* __restrict__ xseq, const float* __restrict__ g,
                            const float* __restrict__ bta,
                            f16* __restrict__ xn, f16* __restrict__ xnrev) {
  int lane = threadIdx.x & 31;
  int row  = blockIdx.x * (blockDim.x >> 5) + (threadIdx.x >> 5);
  if (row >= MROWS) return;
  const float* p = xseq + (size_t)row * 128;
  float v[4], s = 0.f;
#pragma unroll
  for (int j = 0; j < 4; ++j) { v[j] = p[lane + 32 * j]; s += v[j]; }
#pragma unroll
  for (int off = 16; off > 0; off >>= 1) s += __shfl_xor(s, off, 32);
  float mu = s * (1.f / 128.f);
  float q = 0.f;
#pragma unroll
  for (int j = 0; j < 4; ++j) { float d = v[j] - mu; q += d * d; }
#pragma unroll
  for (int off = 16; off > 0; off >>= 1) q += __shfl_xor(q, off, 32);
  float inv = rsqrtf(q * (1.f / 128.f) + 1e-5f);
  int b = row >> 10, l = row & 1023;
  int rrev = (b << 10) + (1023 - l);
#pragma unroll
  for (int j = 0; j < 4; ++j) {
    int c = lane + 32 * j;
    f16 h = (f16)((v[j] - mu) * inv * g[c] + bta[c]);
    xn[(size_t)row * 128 + c]     = h;
    xnrev[(size_t)rrev * 128 + c] = h;
  }
}

// causal depthwise conv(4) + silu on xz[...,:256]
__global__ void k_convsilu(const float* __restrict__ xz, const float* __restrict__ cw,
                           const float* __restrict__ cb,
                           float* __restrict__ xc, f16* __restrict__ xc16) {
  int i = blockIdx.x * blockDim.x + threadIdx.x;
  if (i >= MROWS * DIN) return;
  int d = i & 255, r = i >> 8;
  int l = r & 1023, b = r >> 10;
  float acc = cb[d];
#pragma unroll
  for (int k = 0; k < 4; ++k) {
    int ls = l + k - 3;
    if (ls >= 0) acc += xz[((size_t)((b << 10) + ls)) * 512 + d] * cw[d * 4 + k];
  }
  float v = silu_(acc);
  xc[i] = v;
  xc16[i] = (f16)v;
}

// extract x_dbl[:, :8] into zero-padded f16 [MROWS x 32] for the dt GEMM
__global__ void k_dtin(const float* __restrict__ xdbl, f16* __restrict__ dtin) {
  int i = blockIdx.x * blockDim.x + threadIdx.x;
  if (i >= MROWS * 32) return;
  int c = i & 31, r = i >> 5;
  dtin[i] = (c < DTR) ? (f16)xdbl[(size_t)r * 48 + c] : (f16)0.f;
}

// dt = softplus(dtraw + dt_b), in place
__global__ void k_softplus(float* __restrict__ dt, const float* __restrict__ dtb) {
  int i = blockIdx.x * blockDim.x + threadIdx.x;
  if (i >= MROWS * DIN) return;
  int d = i & 255;
  dt[i] = softplus_(dt[i] + dtb[d]);
}

// selective scan: one block per batch, thread = d-channel, h[16] in registers.
__global__ void k_scan(const float* __restrict__ dt, const float* __restrict__ xc,
                       const float* __restrict__ xdbl, const float* __restrict__ Alog,
                       float* __restrict__ ys) {
  __shared__ float sBC[32];               // B_t (16) then C_t (16)
  int d = threadIdx.x;                    // 0..255
  int b = blockIdx.x;                     // 0..7
  float A[NST], h[NST];
#pragma unroll
  for (int n = 0; n < NST; ++n) { A[n] = -__expf(Alog[d * NST + n]); h[n] = 0.f; }
  for (int l = 0; l < L_; ++l) {
    int r = (b << 10) + l;
    if (threadIdx.x < 32) sBC[threadIdx.x] = xdbl[(size_t)r * 48 + 8 + threadIdx.x];
    __syncthreads();
    float dtv = dt[(size_t)r * DIN + d];
    float xv  = xc[(size_t)r * DIN + d];
    float dx  = dtv * xv;
    float y = 0.f;
#pragma unroll
    for (int n = 0; n < NST; ++n) {
      h[n] = __expf(dtv * A[n]) * h[n] + dx * sBC[n];
      y += h[n] * sBC[16 + n];
    }
    ys[(size_t)r * DIN + d] = y;
    __syncthreads();
  }
}

// (ys + D*xc) * silu(z) -> f16 for out-proj GEMM
__global__ void k_ymul(const float* __restrict__ ys, const float* __restrict__ xc,
                       const float* __restrict__ xz, const float* __restrict__ Dp,
                       f16* __restrict__ ymul) {
  int i = blockIdx.x * blockDim.x + threadIdx.x;
  if (i >= MROWS * DIN) return;
  int d = i & 255, r = i >> 8;
  float y = ys[i] + Dp[d] * xc[i];
  float z = xz[(size_t)r * 512 + 256 + d];
  ymul[i] = (f16)(y * silu_(z));
}

// of + reverse(ob) -> f16 for vim GEMM
__global__ void k_sumfb(const float* __restrict__ of, const float* __restrict__ obr,
                        f16* __restrict__ sum16) {
  int i = blockIdx.x * blockDim.x + threadIdx.x;
  if (i >= MROWS * MAIN_) return;
  int c = i & 127, r = i >> 7;
  int l = r & 1023, b = r >> 10;
  int rr = (b << 10) + (1023 - l);
  sum16[i] = (f16)(of[i] + obr[(size_t)rr * 128 + c]);
}

// x_main = vimout + vim_b + x_seq -> Xout16 cols [0,128)
__global__ void k_vim_post(const float* __restrict__ vimout, const float* __restrict__ vimb,
                           const float* __restrict__ mainseq, f16* __restrict__ Xout16) {
  int i = blockIdx.x * blockDim.x + threadIdx.x;
  if (i >= MROWS * MAIN_) return;
  int c = i & 127, r = i >> 7;
  Xout16[(size_t)r * FUSED + c] = (f16)(vimout[i] + vimb[c] + mainseq[i]);
}

// depthwise 3x3 (ch 0..31) / 5x5 (ch 32..63) + bn + silu -> Xout16 cols [128,192)
__global__ void k_aux_dw(const float* __restrict__ auxin,
                         const float* __restrict__ k3w, const float* __restrict__ k3b,
                         const float* __restrict__ k5w, const float* __restrict__ k5b,
                         const float* __restrict__ g, const float* __restrict__ beta,
                         f16* __restrict__ Xout16) {
  int i = blockIdx.x * blockDim.x + threadIdx.x;
  if (i >= B_ * AUX_ * T2 * F2) return;
  int f = i & 15, t = (i >> 4) & 63, o = (i >> 10) & 63, b = i >> 16;
  const float* src = auxin + (size_t)(b * AUX_ + o) * T2 * F2;
  float acc;
  if (o < 32) {
    acc = k3b[o];
#pragma unroll
    for (int di = -1; di <= 1; ++di)
#pragma unroll
      for (int dj = -1; dj <= 1; ++dj) {
        int tt = t + di, ff = f + dj;
        if (tt >= 0 && tt < T2 && ff >= 0 && ff < F2)
          acc += src[tt * F2 + ff] * k3w[(o * 3 + (di + 1)) * 3 + (dj + 1)];
      }
  } else {
    int oo = o - 32;
    acc = k5b[oo];
#pragma unroll
    for (int di = -2; di <= 2; ++di)
#pragma unroll
      for (int dj = -2; dj <= 2; ++dj) {
        int tt = t + di, ff = f + dj;
        if (tt >= 0 && tt < T2 && ff >= 0 && ff < F2)
          acc += src[tt * F2 + ff] * k5w[(oo * 5 + (di + 2)) * 5 + (dj + 2)];
      }
  }
  float v = silu_(acc * (g[o] * BNS) + beta[o]);
  int r = (b * T2 + t) * F2 + f;
  Xout16[(size_t)r * FUSED + 128 + o] = (f16)v;
}

// x_fused = x_in + bn(fus + fus_b) -> f16 for rec GEMM
__global__ void k_fus_post(const float* __restrict__ Fus, const float* __restrict__ fb,
                           const float* __restrict__ fg, const float* __restrict__ fbeta,
                           const float* __restrict__ mainseq, const float* __restrict__ auxin,
                           f16* __restrict__ Xf16) {
  int i = blockIdx.x * blockDim.x + threadIdx.x;
  if (i >= MROWS * FUSED) return;
  int c = i % FUSED, r = i / FUSED;
  float v = (Fus[i] + fb[c]) * (fg[c] * BNS) + fbeta[c];
  float xin;
  if (c < 128) {
    xin = mainseq[(size_t)r * 128 + c];
  } else {
    int o = c - 128;
    int f = r & 15, t = (r >> 4) & 63, b = r >> 10;
    xin = auxin[((size_t)(b * AUX_ + o) * T2 + t) * F2 + f];
  }
  Xf16[i] = (f16)(xin + v);
}

// out = relu((R + rec_b)*rec_g*BNS + rec_beta) + x  with pixel-shuffle layout
__global__ void k_rec_post(const float* __restrict__ R, const float* __restrict__ rb,
                           const float* __restrict__ rg, const float* __restrict__ rbeta,
                           const float* __restrict__ x, float* __restrict__ out) {
  int i = blockIdx.x * blockDim.x + threadIdx.x;
  if (i >= B_ * CIN * T_ * F_) return;
  int jf = i & 31;
  int it = (i >> 5) & 127;
  int o  = (i >> 12) & 63;
  int b  = i >> 18;
  int k = it & 1, ii = it >> 1;
  int l = jf & 1, jj = jf >> 1;
  int r = (b * T2 + ii) * F2 + jj;
  float v = R[(size_t)r * 256 + (o * 4 + k * 2 + l)];
  v = (v + rb[o]) * (rg[o] * BNS) + rbeta[o];
  out[i] = fmaxf(v, 0.f) + x[i];
}

// ---------------------------------------------------------------------------
// Host launcher
// ---------------------------------------------------------------------------
template<int K, int N>
static inline void launch_gemm(const f16* A, const f16* Bsw, float* C, hipStream_t s) {
  constexpr int waves  = (MROWS >> 6) * (N >> 4);  // 4 M-tiles per wave
  constexpr int blocks = waves / 4;                // 4 waves (128 threads) per block
  gemm_sw<K, N><<<dim3(blocks), dim3(128), 0, s>>>(A, Bsw, C);
}

#define EW(n) dim3(((n) + 255) / 256), dim3(256), 0, stream

extern "C" void kernel_launch(void* const* d_in, const int* in_sizes, int n_in,
                              void* d_out, int out_size, void* d_ws, size_t ws_size,
                              hipStream_t stream) {
  (void)in_sizes; (void)n_in; (void)out_size; (void)ws_size;

  const float* x         = (const float*)d_in[0];
  const float* main_w    = (const float*)d_in[1];
  const float* main_b    = (const float*)d_in[2];
  const float* main_g    = (const float*)d_in[3];
  const float* main_beta = (const float*)d_in[4];
  const float* aux_w     = (const float*)d_in[5];
  const float* aux_b     = (const float*)d_in[6];
  const float* aux_g     = (const float*)d_in[7];
  const float* aux_beta  = (const float*)d_in[8];
  const float* ln_g      = (const float*)d_in[9];
  const float* ln_b      = (const float*)d_in[10];
  const float* vim_w     = (const float*)d_in[11];
  const float* vim_b     = (const float*)d_in[12];
  const float* mk3_w     = (const float*)d_in[13];
  const float* mk3_b     = (const float*)d_in[14];
  const float* mk5_w     = (const float*)d_in[15];
  const float* mk5_b     = (const float*)d_in[16];
  const float* mk_g      = (const float*)d_in[17];
  const float* mk_beta   = (const float*)d_in[18];
  const float* fus_w     = (const float*)d_in[19];
  const float* fus_b     = (const float*)d_in[20];
  const float* fus_g     = (const float*)d_in[21];
  const float* fus_beta  = (const float*)d_in[22];
  const float* rec_w     = (const float*)d_in[23];
  const float* rec_b     = (const float*)d_in[24];
  const float* rec_g     = (const float*)d_in[25];
  const float* rec_beta  = (const float*)d_in[26];
  // per-direction params: f_* at 27..35, b_* at 36..44
  const float* in_w[2]    = { (const float*)d_in[27], (const float*)d_in[36] };
  const float* conv_w[2]  = { (const float*)d_in[28], (const float*)d_in[37] };
  const float* conv_b[2]  = { (const float*)d_in[29], (const float*)d_in[38] };
  const float* xproj_w[2] = { (const float*)d_in[30], (const float*)d_in[39] };
  const float* dt_w[2]    = { (const float*)d_in[31], (const float*)d_in[40] };
  const float* dt_b[2]    = { (const float*)d_in[32], (const float*)d_in[41] };
  const float* Alog[2]    = { (const float*)d_in[33], (const float*)d_in[42] };
  const float* Dpar[2]    = { (const float*)d_in[34], (const float*)d_in[43] };
  const float* out_w[2]   = { (const float*)d_in[35], (const float*)d_in[44] };

  float* out = (float*)d_out;

  // ---- workspace carve-out ----
  char* base = (char*)d_ws;
  size_t off = 0;
  auto alloc = [&](size_t bytes) -> void* {
    void* p = base + off;
    off += (bytes + 255) & ~(size_t)255;
    return p;
  };
  f16* Wt_main   = (f16*)alloc(128 * 128 * 2);
  f16* Wt_aux    = (f16*)alloc(128 * 64 * 2);
  f16* Wt_in[2]  = { (f16*)alloc(128 * 512 * 2), (f16*)alloc(128 * 512 * 2) };
  f16* Wt_xp[2]  = { (f16*)alloc(256 * 48 * 2),  (f16*)alloc(256 * 48 * 2)  };
  f16* Wt_dt[2]  = { (f16*)alloc(32 * 256 * 2),  (f16*)alloc(32 * 256 * 2)  };
  f16* Wt_out[2] = { (f16*)alloc(256 * 128 * 2), (f16*)alloc(256 * 128 * 2) };
  f16* Wt_vim    = (f16*)alloc(128 * 128 * 2);
  f16* Wt_fus    = (f16*)alloc(192 * 192 * 2);
  f16* Wt_rec    = (f16*)alloc(192 * 256 * 2);

  f16*   A_low   = (f16*)  alloc((size_t)MROWS * 128 * 2);
  f16*   A_high  = (f16*)  alloc((size_t)MROWS * 128 * 2);
  float* Ymain   = (float*)alloc((size_t)MROWS * 128 * 4);
  float* Yaux    = (float*)alloc((size_t)MROWS * 64 * 4);
  float* mainseq = (float*)alloc((size_t)MROWS * 128 * 4);
  float* auxin   = (float*)alloc((size_t)B_ * AUX_ * T2 * F2 * 4);
  f16*   xn16    = (f16*)  alloc((size_t)MROWS * 128 * 2);
  f16*   xnrev16 = (f16*)  alloc((size_t)MROWS * 128 * 2);
  float* XZ      = (float*)alloc((size_t)MROWS * 512 * 4);
  float* xc      = (float*)alloc((size_t)MROWS * 256 * 4);
  f16*   xc16    = (f16*)  alloc((size_t)MROWS * 256 * 2);
  float* xdbl    = (float*)alloc((size_t)MROWS * 48 * 4);
  f16*   dtin16  = (f16*)  alloc((size_t)MROWS * 32 * 2);
  float* dtbuf   = (float*)alloc((size_t)MROWS * 256 * 4);
  float* ysbuf   = (float*)alloc((size_t)MROWS * 256 * 4);
  f16*   ymul16  = (f16*)  alloc((size_t)MROWS * 256 * 2);
  float* odir[2] = { (float*)alloc((size_t)MROWS * 128 * 4),
                     (float*)alloc((size_t)MROWS * 128 * 4) };
  f16*   sum16   = (f16*)  alloc((size_t)MROWS * 128 * 2);
  float* vimout  = (float*)alloc((size_t)MROWS * 128 * 4);
  f16*   Xout16  = (f16*)  alloc((size_t)MROWS * 192 * 2);
  float* Fus     = (float*)alloc((size_t)MROWS * 192 * 4);
  f16*   Xf16    = (f16*)  alloc((size_t)MROWS * 192 * 2);
  float* Rec     = (float*)alloc((size_t)MROWS * 256 * 4);

  // ---- weight prep: transpose/pad + fragment-swizzle (tiny) ----
  k_swizzle_w<<<EW(128 * 128)>>>(main_w, Wt_main, 128, 128, 128, 128, 1);
  k_swizzle_w<<<EW(64 * 128)>>>(aux_w, Wt_aux, 64, 128, 64, 128, 1);
  for (int p = 0; p < 2; ++p) {
    k_swizzle_w<<<EW(512 * 128)>>>(in_w[p],   Wt_in[p],  512, 128, 512, 128, 1);
    k_swizzle_w<<<EW(48 * 256)>>>(xproj_w[p], Wt_xp[p],  40,  256, 48,  256, 1);
    k_swizzle_w<<<EW(256 * 32)>>>(dt_w[p],    Wt_dt[p],  256, 8,   256, 32,  1);
    k_swizzle_w<<<EW(128 * 256)>>>(out_w[p],  Wt_out[p], 128, 256, 128, 256, 1);
  }
  k_swizzle_w<<<EW(128 * 128)>>>(vim_w, Wt_vim, 128, 128, 128, 128, 1);
  k_swizzle_w<<<EW(192 * 192)>>>(fus_w, Wt_fus, 192, 192, 192, 192, 1);
  k_swizzle_w<<<EW(256 * 192)>>>(rec_w, Wt_rec, 256, 192, 256, 192, 0);

  // ---- wavelet + patch projections ----
  k_wavelet<<<EW(MROWS * 128)>>>(x, A_low, A_high);
  launch_gemm<128, 128>(A_low,  Wt_main, Ymain, stream);
  launch_gemm<128, 64>(A_high, Wt_aux,  Yaux,  stream);
  k_patch_post_main<<<EW(MROWS * 128)>>>(Ymain, main_b, main_g, main_beta, mainseq);
  k_patch_post_aux<<<EW(MROWS * 64)>>>(Yaux, aux_b, aux_g, aux_beta, auxin);

  // ---- LayerNorm (forward + reversed copies) ----
  k_layernorm<<<dim3(MROWS / 8), dim3(256), 0, stream>>>(mainseq, ln_g, ln_b, xn16, xnrev16);

  // ---- bidirectional Mamba ----
  for (int p = 0; p < 2; ++p) {
    const f16* xin = (p == 0) ? xn16 : xnrev16;
    launch_gemm<128, 512>(xin, Wt_in[p], XZ, stream);
    k_convsilu<<<EW(MROWS * 256)>>>(XZ, conv_w[p], conv_b[p], xc, xc16);
    launch_gemm<256, 48>(xc16, Wt_xp[p], xdbl, stream);
    k_dtin<<<EW(MROWS * 32)>>>(xdbl, dtin16);
    launch_gemm<32, 256>(dtin16, Wt_dt[p], dtbuf, stream);
    k_softplus<<<EW(MROWS * 256)>>>(dtbuf, dt_b[p]);
    k_scan<<<dim3(B_), dim3(256), 0, stream>>>(dtbuf, xc, xdbl, Alog[p], ysbuf);
    k_ymul<<<EW(MROWS * 256)>>>(ysbuf, xc, XZ, Dpar[p], ymul16);
    launch_gemm<256, 128>(ymul16, Wt_out[p], odir[p], stream);
  }

  // ---- vim projection + residual ----
  k_sumfb<<<EW(MROWS * 128)>>>(odir[0], odir[1], sum16);
  launch_gemm<128, 128>(sum16, Wt_vim, vimout, stream);
  k_vim_post<<<EW(MROWS * 128)>>>(vimout, vim_b, mainseq, Xout16);

  // ---- aux depthwise branch ----
  k_aux_dw<<<EW(B_ * AUX_ * T2 * F2)>>>(auxin, mk3_w, mk3_b, mk5_w, mk5_b,
                                        mk_g, mk_beta, Xout16);

  // ---- fusion ----
  launch_gemm<192, 192>(Xout16, Wt_fus, Fus, stream);
  k_fus_post<<<EW(MROWS * 192)>>>(Fus, fus_b, fus_g, fus_beta, mainseq, auxin, Xf16);

  // ---- reconstruction + residual ----
  launch_gemm<192, 256>(Xf16, Wt_rec, Rec, stream);
  k_rec_post<<<EW(B_ * CIN * T_ * F_)>>>(Rec, rec_b, rec_g, rec_beta, x, out);
}
